// MultiheadAttention_1529008357598
// MI455X (gfx1250) — matrix operand order repaired and checked by reading, not compiled, not run on hardware
//
#include <hip/hip_runtime.h>

// ---------------------------------------------------------------------------
// MultiheadAttention (per-head linear projection + l2norm + softmax + V mix)
// CDNA5 / gfx1250, wave32. All GEMMs via v_wmma_f32_16x16x32_f16.
// Attention: 64-q-row workgroup tile, non-redundant score producers,
// double-buffered async (GLOBAL_LOAD_ASYNC_TO_LDS_B128) V staging.
// ---------------------------------------------------------------------------

typedef __attribute__((ext_vector_type(16))) _Float16 v16h;
typedef __attribute__((ext_vector_type(8)))  _Float16 v8h;
typedef __attribute__((ext_vector_type(4)))  _Float16 v4h;
typedef __attribute__((ext_vector_type(8)))  float    v8f;
typedef __attribute__((ext_vector_type(4)))  int      v4i;

#define LQ   1024
#define LK   1024
#define BB   16
#define FF   512
#define HH   8
#define KD   64            // per-head key feature dim
#define NROW (LQ * BB)     // 16384 flattened (q,b) rows
#define PC   (HH * KD)     // 512 projection cols
#define TEMP 30.0f
#define VPAD 72            // padded row pitch (f16) for LDS tiles
#define NT   (LK / 64)     // 16 s-tiles

#if __has_builtin(__builtin_amdgcn_global_load_async_to_lds_b128) && \
    __has_builtin(__builtin_amdgcn_s_wait_asynccnt)
#define HAVE_ASYNC 1
#define ASYNC_COPY_B128(dst, src)                                         \
    __builtin_amdgcn_global_load_async_to_lds_b128(                       \
        (__attribute__((address_space(1))) v4i*)(src),                    \
        (__attribute__((address_space(3))) v4i*)(dst), 0, 0)
#define WAIT_ASYNC(n) __builtin_amdgcn_s_wait_asynccnt(n)
#else
#define HAVE_ASYNC 0
#define ASYNC_COPY_B128(dst, src) (*(v8h*)(dst) = *(const v8h*)(src))
#define WAIT_ASYNC(n) ((void)0)
#endif

// ---------------------------------------------------------------------------
// Stage 0a: f32 -> f16 bulk convert
// ---------------------------------------------------------------------------
__global__ void cvt_f16_kernel(const float* __restrict__ in,
                               _Float16* __restrict__ out, int n4) {
    int i = blockIdx.x * blockDim.x + threadIdx.x;
    if (i < n4) {
        float4 x = ((const float4*)in)[i];
        v4h y;
        y[0] = (_Float16)x.x; y[1] = (_Float16)x.y;
        y[2] = (_Float16)x.z; y[3] = (_Float16)x.w;
        ((v4h*)out)[i] = y;
    }
}

// ---------------------------------------------------------------------------
// Stage 0b: value [s][b][f] f32 -> Vt [b][f][s] f16 (so attn*V B-fragments
// have their K-dim (=s) contiguous: one 32B load per lane per fragment).
// ---------------------------------------------------------------------------
__global__ __launch_bounds__(256)
void transpose_v_f16(const float* __restrict__ v, _Float16* __restrict__ vt) {
    __shared__ _Float16 t[32][40];
    const int b  = blockIdx.z;
    const int s0 = blockIdx.x * 32, f0 = blockIdx.y * 32;
    const int tx = threadIdx.x, ty = threadIdx.y;   // block (32, 8)
#pragma unroll
    for (int i = 0; i < 32; i += 8) {
        float x = v[((size_t)(s0 + ty + i) * BB + b) * FF + f0 + tx];
        t[ty + i][tx] = (_Float16)x;
    }
    __syncthreads();
#pragma unroll
    for (int i = 0; i < 32; i += 8) {
        vt[((size_t)b * FF + f0 + ty + i) * LK + s0 + tx] = t[tx][ty + i];
    }
}

// ---------------------------------------------------------------------------
// Stage 1: Y = l2norm_per_head(X @ W^T + bias), f16 out.
// Wave computes a 16x64 tile; 64-col tile == one head, so the L2 norm is
// in-register (shfl_xor over the 16-lane C/D half-groups).
// ---------------------------------------------------------------------------
__global__ __launch_bounds__(256)
void proj_norm(const _Float16* __restrict__ X, const _Float16* __restrict__ W,
               const float* __restrict__ bias, _Float16* __restrict__ Y) {
    const int lane = threadIdx.x & 31;
    const int wave = threadIdx.x >> 5;
    const int m    = lane & 15;     // A-row / B-col / D-col index
    const int kh   = lane >> 4;     // k-half select (wave32 WMMA layouts)
    const int r0   = blockIdx.x * 128 + wave * 16;
    const int c0   = blockIdx.y * 64;

    v8f acc[4] = {};
    const _Float16* xrow = X + (size_t)(r0 + m) * FF;

    for (int kk = 0; kk < FF; kk += 32) {
        v8h lo = *(const v8h*)(xrow + kk + kh * 16);
        v8h hi = *(const v8h*)(xrow + kk + kh * 16 + 8);
        v16h a;
#pragma unroll
        for (int i = 0; i < 8; ++i) { a[i] = lo[i]; a[i + 8] = hi[i]; }
#pragma unroll
        for (int j = 0; j < 4; ++j) {
            const _Float16* wrow =
                W + (size_t)(c0 + j * 16 + m) * FF + kk + kh * 16;
            v16h bf = *(const v16h*)wrow;
            acc[j] = __builtin_amdgcn_wmma_f32_16x16x32_f16(
                false, a, false, bf, (short)0, acc[j], false, false);
        }
    }

    float ss[8];
#pragma unroll
    for (int rr = 0; rr < 8; ++rr) ss[rr] = 0.0f;
#pragma unroll
    for (int j = 0; j < 4; ++j) {
        float bj = bias[c0 + j * 16 + m];
#pragma unroll
        for (int rr = 0; rr < 8; ++rr) {
            acc[j][rr] += bj;
            ss[rr] += acc[j][rr] * acc[j][rr];
        }
    }
#pragma unroll
    for (int rr = 0; rr < 8; ++rr) {
        float s = ss[rr];
        s += __shfl_xor(s, 1, 32);
        s += __shfl_xor(s, 2, 32);
        s += __shfl_xor(s, 4, 32);
        s += __shfl_xor(s, 8, 32);
        ss[rr] = 1.0f / fmaxf(sqrtf(s), 1e-12f);
    }
#pragma unroll
    for (int j = 0; j < 4; ++j) {
#pragma unroll
        for (int rr = 0; rr < 8; ++rr) {
            Y[(size_t)(r0 + rr + 8 * kh) * PC + c0 + j * 16 + m] =
                (_Float16)(acc[j][rr] * ss[rr]);
        }
    }
}

// ---------------------------------------------------------------------------
// Stage 2: streaming (flash) attention, 64 q rows per workgroup.
//   waves 0-3  : produce scores+softmax+P for q-group w (16 rows) -- once.
//   all 8 waves: consume as (qg = w>>1, fc = w&1): 16 q rows x 256 F cols,
//                accumulators = 16 v8f C-tiles (128 VGPRs).
// V tile (64 s x 512 f) is staged to LDS with async B128 copies, double
// buffered: the copy of tile s+1 overlaps the score WMMAs of tile s.
// ---------------------------------------------------------------------------
__global__ __launch_bounds__(256)
void attn_kernel(const _Float16* __restrict__ Qn,  // [NROW, PC] normalized
                 const _Float16* __restrict__ Kn,  // [NROW, PC] normalized
                 const _Float16* __restrict__ Vt,  // [B][FF][LK] f16
                 float* __restrict__ out) {        // [LQ][B][HH*FF] f32
    __shared__ __attribute__((aligned(16))) _Float16 Vs[2][FF * VPAD];
    __shared__ __attribute__((aligned(16))) _Float16 Pl[4][16 * VPAD];
    __shared__ __attribute__((aligned(32))) float    Al[4 * 16];
    __shared__ __attribute__((aligned(32))) float    Ll[4 * 16];

    const int tid  = threadIdx.x;
    const int lane = tid & 31;
    const int wave = tid >> 5;
    const int m    = lane & 15;
    const int kh   = lane >> 4;
    const int q0   = blockIdx.x * 64;
    const int h    = blockIdx.y;
    const int b    = blockIdx.z;
    const int qgc  = wave >> 1;        // consumed q-group
    const int fc   = (wave & 1) * 256; // consumed F-chunk base

    const size_t rstride = (size_t)BB * PC;   // 8192 f16 between L steps
    const _Float16* vtb  = Vt + (size_t)b * FF * LK;

    // ---- producer state (waves 0-3, q-group = wave) ----
    v16h a0, a1;
    float mrow[8], lrow[8];
    const _Float16* kbase = Kn + (size_t)b * PC + h * KD;
    if (wave < 4) {
        const _Float16* qrow =
            Qn + (size_t)(q0 + wave * 16 + m) * rstride + (size_t)b * PC + h * KD;
        v8h lo = *(const v8h*)(qrow + kh * 16);
        v8h hi = *(const v8h*)(qrow + kh * 16 + 8);
#pragma unroll
        for (int i = 0; i < 8; ++i) { a0[i] = lo[i]; a0[i + 8] = hi[i]; }
        lo = *(const v8h*)(qrow + 32 + kh * 16);
        hi = *(const v8h*)(qrow + 32 + kh * 16 + 8);
#pragma unroll
        for (int i = 0; i < 8; ++i) { a1[i] = lo[i]; a1[i + 8] = hi[i]; }
    }
#pragma unroll
    for (int rr = 0; rr < 8; ++rr) { mrow[rr] = -3.0e38f; lrow[rr] = 0.0f; }

    // ---- consumer accumulators: 16 q rows x 256 f cols ----
    v8f o[16];
#pragma unroll
    for (int j = 0; j < 16; ++j) o[j] = (v8f){};

    // ---- stage V tile s0 into Vs[buf] : 4096 B128 chunks / 256 threads ----
    auto stage_v = [&](int buf, int s0) {
#pragma unroll
        for (int i = 0; i < 16; ++i) {
            int c  = tid + i * 256;        // chunk id
            int f  = c >> 3;               // f row (0..511)
            int s8 = (c & 7) * 8;          // s sub-offset (f16 units)
            const _Float16* g = vtb + (size_t)f * LK + s0 + s8;
            _Float16*       l = &Vs[buf][f * VPAD + s8];
            ASYNC_COPY_B128(l, g);
        }
    };

    stage_v(0, 0);                         // preload tile 0

    for (int st = 0; st < NT; ++st) {
        const int s0  = st * 64;
        const int buf = st & 1;

        if (st + 1 < NT) stage_v(buf ^ 1, s0 + 64);   // overlap with scores

        // ---------------- producers: S = TEMP * Q K^T + online softmax ----
        if (wave < 4) {
            v8f sc[4];
#pragma unroll
            for (int j = 0; j < 4; ++j) {
                const _Float16* krow =
                    kbase + (size_t)(s0 + j * 16 + m) * rstride + kh * 16;
                v16h b0 = *(const v16h*)(krow);
                v16h b1 = *(const v16h*)(krow + 32);
                v8f z = {};
                z = __builtin_amdgcn_wmma_f32_16x16x32_f16(
                    false, a0, false, b0, (short)0, z, false, false);
                z = __builtin_amdgcn_wmma_f32_16x16x32_f16(
                    false, a1, false, b1, (short)0, z, false, false);
                sc[j] = z;
            }
            if (st + 1 < NT)   // pull next K tile toward L2
                __builtin_prefetch(kbase + (size_t)(s0 + 64 + m) * rstride, 0, 1);
#pragma unroll
            for (int rr = 0; rr < 8; ++rr) {
                float tm = -3.0e38f;
#pragma unroll
                for (int j = 0; j < 4; ++j) {
                    sc[j][rr] *= TEMP;
                    tm = fmaxf(tm, sc[j][rr]);
                }
                tm = fmaxf(tm, __shfl_xor(tm, 1, 32));
                tm = fmaxf(tm, __shfl_xor(tm, 2, 32));
                tm = fmaxf(tm, __shfl_xor(tm, 4, 32));
                tm = fmaxf(tm, __shfl_xor(tm, 8, 32));
                float mnew  = fmaxf(mrow[rr], tm);
                float alpha = __expf(mrow[rr] - mnew);
                mrow[rr] = mnew;
                float rs = 0.0f;
#pragma unroll
                for (int j = 0; j < 4; ++j) {
                    float pv = __expf(sc[j][rr] - mnew);
                    sc[j][rr] = pv;
                    rs += pv;
                }
                rs += __shfl_xor(rs, 1, 32);
                rs += __shfl_xor(rs, 2, 32);
                rs += __shfl_xor(rs, 4, 32);
                rs += __shfl_xor(rs, 8, 32);
                lrow[rr] = lrow[rr] * alpha + rs;
                if (m == 0) Al[wave * 16 + kh * 8 + rr] = alpha;
            }
#pragma unroll
            for (int j = 0; j < 4; ++j)
#pragma unroll
                for (int rr = 0; rr < 8; ++rr)
                    Pl[wave][(rr + 8 * kh) * VPAD + j * 16 + m] =
                        (_Float16)sc[j][rr];
        }

        if (st + 1 < NT) { WAIT_ASYNC(16); } else { WAIT_ASYNC(0); }
        __syncthreads();   // V(buf), P, alpha all visible

        // ---------------- consumers: O = O*alpha + P @ V ------------------
        v8f av = *(const v8f*)(&Al[qgc * 16 + kh * 8]);
        v16h pa0, pa1;
        {
            const _Float16* prow = &Pl[qgc][m * VPAD];
            v8h lo = *(const v8h*)(prow + kh * 16);
            v8h hi = *(const v8h*)(prow + kh * 16 + 8);
#pragma unroll
            for (int i = 0; i < 8; ++i) { pa0[i] = lo[i]; pa0[i + 8] = hi[i]; }
            lo = *(const v8h*)(prow + 32 + kh * 16);
            hi = *(const v8h*)(prow + 32 + kh * 16 + 8);
#pragma unroll
            for (int i = 0; i < 8; ++i) { pa1[i] = lo[i]; pa1[i + 8] = hi[i]; }
        }
#pragma unroll
        for (int j = 0; j < 16; ++j) {
#pragma unroll
            for (int rr = 0; rr < 8; ++rr) o[j][rr] *= av[rr];
            const _Float16* vrow = &Vs[buf][(fc + j * 16 + m) * VPAD + kh * 16];
            v16h b0 = *(const v16h*)(vrow);
            v16h b1 = *(const v16h*)(vrow + 32);
            o[j] = __builtin_amdgcn_wmma_f32_16x16x32_f16(
                false, pa0, false, b0, (short)0, o[j], false, false);
            o[j] = __builtin_amdgcn_wmma_f32_16x16x32_f16(
                false, pa1, false, b1, (short)0, o[j], false, false);
        }
        __syncthreads();   // P/alpha reuse next tile
    }

    // ---- final 1/l, then store [q][b][h*FF + f] f32 ----
    if (wave < 4 && m == 0) {
#pragma unroll
        for (int rr = 0; rr < 8; ++rr)
            Ll[wave * 16 + kh * 8 + rr] = 1.0f / lrow[rr];
    }
    __syncthreads();
    v8f lv = *(const v8f*)(&Ll[qgc * 16 + kh * 8]);
#pragma unroll
    for (int j = 0; j < 16; ++j) {
#pragma unroll
        for (int rr = 0; rr < 8; ++rr) {
            out[((size_t)(q0 + qgc * 16 + rr + 8 * kh) * BB + b) * (HH * FF) +
                h * FF + fc + j * 16 + m] = o[j][rr] * lv[rr];
        }
    }
}

// ---------------------------------------------------------------------------
extern "C" void kernel_launch(void* const* d_in, const int* in_sizes, int n_in,
                              void* d_out, int out_size, void* d_ws, size_t ws_size,
                              hipStream_t stream) {
    const float* q  = (const float*)d_in[0];
    const float* k  = (const float*)d_in[1];
    const float* v  = (const float*)d_in[2];
    const float* Wk = (const float*)d_in[3];
    const float* bk = (const float*)d_in[4];
    float* out = (float*)d_out;

    _Float16* p   = (_Float16*)d_ws;
    _Float16* qh  = p; p += (size_t)NROW * FF;    // query f16
    _Float16* khb = p; p += (size_t)NROW * FF;    // key f16
    _Float16* wh  = p; p += (size_t)PC * FF;      // Wk f16
    _Float16* wqn = p; p += (size_t)NROW * PC;    // normalized wq f16
    _Float16* wkn = p; p += (size_t)NROW * PC;    // normalized wk f16
    _Float16* vt  = p; p += (size_t)BB * FF * LK; // V^T f16 [b][f][s]

    {
        int n4 = NROW * FF / 4;
        cvt_f16_kernel<<<(n4 + 255) / 256, 256, 0, stream>>>(q, qh, n4);
        cvt_f16_kernel<<<(n4 + 255) / 256, 256, 0, stream>>>(k, khb, n4);
        int w4 = PC * FF / 4;
        cvt_f16_kernel<<<(w4 + 255) / 256, 256, 0, stream>>>(Wk, wh, w4);
    }
    transpose_v_f16<<<dim3(LK / 32, FF / 32, BB), dim3(32, 8), 0, stream>>>(v, vt);

    proj_norm<<<dim3(NROW / 128, PC / 64), 256, 0, stream>>>(qh, wh, bk, wqn);
    proj_norm<<<dim3(NROW / 128, PC / 64), 256, 0, stream>>>(khb, wh, bk, wkn);

    attn_kernel<<<dim3(LQ / 64, HH, BB), 256, 0, stream>>>(wqn, wkn, vt, out);
}
